// GRUDDecoder_66520453481163
// MI455X (gfx1250) — matrix operand order repaired and checked by reading, not compiled, not run on hardware
//
#include <hip/hip_runtime.h>
#include <math.h>

#define T_STEPS 100
#define BN 512
#define HN 512
#define ON 512

typedef __attribute__((ext_vector_type(16))) __bf16   v16bf;
typedef __attribute__((ext_vector_type(8)))  float    v8f;
typedef __attribute__((ext_vector_type(4)))  unsigned int u32x4;

union FragBF { v16bf v; u32x4 q[2]; };

__device__ __forceinline__ unsigned short f2bf(float f) {
  unsigned u = __builtin_bit_cast(unsigned, f);
  u += 0x7FFFu + ((u >> 16) & 1u);          // round-to-nearest-even
  return (unsigned short)(u >> 16);
}

// A-matrix 16x32 bf16 fragment (ISA 7.12.2): lanes 0-15 -> K {0..7,16..23},
// lanes 16-31 -> K {8..15,24..31}; M = lane&15. Row-major source, ld in elems.
__device__ __forceinline__ v16bf load_a_frag(const unsigned short* tile, int ld, int lane) {
  int m  = lane & 15;
  int kk = (lane >> 4) << 3;
  const unsigned short* p = tile + m * ld + kk;
  FragBF f;
  f.q[0] = *(const u32x4*)(p);
  f.q[1] = *(const u32x4*)(p + 16);
  return f.v;
}

// B-matrix 32x16 bf16 fragment: lane = N (0..15), lanes 0-15 -> K {0..7,16..23},
// lanes 16-31 -> K {8..15,24..31}; source is the TRANSPOSED weight (N x K row-major).
__device__ __forceinline__ v16bf load_b_frag(const unsigned short* tileT, int ld, int lane) {
  int n  = lane & 15;
  int kh = (lane >> 4) << 3;
  const unsigned short* p = tileT + n * ld + kh;
  FragBF f;
  f.q[0] = *(const u32x4*)(p);
  f.q[1] = *(const u32x4*)(p + 16);
  return f.v;
}

// ---------- one-time prep: bf16 transposed weights + column sums ----------
__global__ void prep_kernel(const float* Wz, const float* Wr, const float* Wh,
                            const float* Wgh, const float* Wlin,
                            const float* bz, const float* br, const float* bh,
                            unsigned short* WzT, unsigned short* WrT, unsigned short* WhT,
                            unsigned short* WlT, float* S, float* colz, float* colr, float* colh) {
  const int NTR = HN * HN;
  int idx = blockIdx.x * blockDim.x + threadIdx.x;
  const int total = 4 * NTR + 4 * HN;
  for (; idx < total; idx += gridDim.x * blockDim.x) {
    if (idx < 3 * NTR) {
      int m = idx / NTR;
      int e = idx % NTR;
      int n = e >> 9, k = e & 511;
      const float* W = (m == 0) ? Wz : ((m == 1) ? Wr : Wh);
      unsigned short* D = (m == 0) ? WzT : ((m == 1) ? WrT : WhT);
      D[e] = f2bf(W[(HN + k) * HN + n]);          // h-block, transposed
    } else if (idx < 4 * NTR) {
      int e = idx - 3 * NTR;
      int o = e >> 9, k = e & 511;
      WlT[e] = f2bf(Wlin[k * ON + o]);
    } else {
      int j = idx - 4 * NTR;
      int which = j >> 9, n = j & 511;
      float s = 0.f;
      if (which == 0) {
        for (int k = 0; k < HN; ++k) s += Wgh[k * HN + n];
        S[n] = s;                                  // colsum(Wgh): rank-1 d@Wgh
      } else {
        const float* W  = (which == 1) ? Wz : ((which == 2) ? Wr : Wh);
        const float* bb = (which == 1) ? bz : ((which == 2) ? br : bh);
        for (int k = 0; k < HN; ++k) s += W[(2 * HN + k) * HN + n];   // ones (mask) block
        float* dst = (which == 1) ? colz : ((which == 2) ? colr : colh);
        dst[n] = s + bb[n];
      }
    }
  }
}

// ---------- one-time: base_{z,r,h} = C @ W_x + bias + mask-colsum (exact f32) ----------
__global__ void base_kernel(const float* C, const float* Wz, const float* Wr, const float* Wh,
                            const float* colz, const float* colr, const float* colh,
                            float* base_z, float* base_r, float* base_h) {
  int idx = blockIdx.x * blockDim.x + threadIdx.x;   // BN*HN threads
  int b = idx >> 9, n = idx & 511;
  float az = 0.f, ar = 0.f, ah = 0.f;
  for (int k = 0; k < HN; ++k) {
    float c = C[b * HN + k];
    az += c * Wz[k * HN + n];
    ar += c * Wr[k * HN + n];
    ah += c * Wh[k * HN + n];
  }
  base_z[idx] = az + colz[n];
  base_r[idx] = ar + colr[n];
  base_h[idx] = ah + colh[n];
}

// ---------- persistent recurrence: 16 batch rows per WG, full T loop in-kernel ----------
// 512 threads = 16 waves; wave -> 32 output columns (2 N-tiles). Double-buffered
// fragment prefetch hides L2 latency behind WMMAs on the sequential chain.
__global__ __launch_bounds__(512) void grud_recur(
    const float* __restrict__ t_in, const float* __restrict__ bgh_g, const float* __restrict__ Sg,
    const float* __restrict__ base_z, const float* __restrict__ base_r, const float* __restrict__ base_h,
    const unsigned short* __restrict__ WzT, const unsigned short* __restrict__ WrT,
    const unsigned short* __restrict__ WhT, unsigned short* __restrict__ hs_bf) {
  __shared__ float          sh_h[16][HN];            // current hidden (f32)
  __shared__ unsigned short sh_a[16][HN];            // bf16 A-operand staging
  __shared__ float          sh_S[HN];
  __shared__ float          sh_bgh[HN];
  __shared__ float          sh_dt[16];

  const int tid  = threadIdx.x;
  const int lane = tid & 31;
  const int wave = tid >> 5;
  const int b0   = blockIdx.x * 16;
  const int nc0  = wave * 32;                        // 2 N-tiles per wave
  const int fm   = (lane >> 4) << 3;                 // frag M base (+j)
  const int fn   = lane & 15;                        // frag N

  for (int i = tid; i < HN; i += 512) { sh_S[i] = Sg[i]; sh_bgh[i] = bgh_g[i]; }
  for (int i = tid; i < 16 * HN; i += 512) sh_h[i >> 9][i & 511] = 0.f;

  for (int t = 0; t < T_STEPS; ++t) {
    __syncthreads();
    if (tid < 16)
      sh_dt[tid] = (t == 0) ? 0.f
                 : (t_in[t * BN + b0 + tid] - t_in[(t - 1) * BN + b0 + tid]);
    __syncthreads();
    // phase 1: gamma_h (rank-1 d@Wgh) and hidden decay; stage bf16 A
    for (int i = tid; i < 16 * HN; i += 512) {
      int m = i >> 9, n = i & 511;
      float g  = __expf(-fmaxf(sh_dt[m] * sh_S[n] + sh_bgh[n], 0.f));
      float hd = g * sh_h[m][n];
      sh_h[m][n] = hd;
      sh_a[m][n] = f2bf(hd);
    }
    __syncthreads();

    // phase 2: z,r gates — two fused GEMMs sharing A fragments, SW-pipelined
    v8f accz[2], accr[2];
#pragma unroll
    for (int i = 0; i < 2; ++i) {
      int ng = nc0 + i * 16 + fn;
#pragma unroll
      for (int j = 0; j < 8; ++j) {
        int off = (b0 + fm + j) * HN + ng;
        accz[i][j] = base_z[off];
        accr[i][j] = base_r[off];
      }
    }
    {
      v16bf aC = load_a_frag(&sh_a[0][0], HN, lane);
      v16bf bzC[2], brC[2];
#pragma unroll
      for (int i = 0; i < 2; ++i) {
        bzC[i] = load_b_frag(WzT + (nc0 + i * 16) * HN, HN, lane);
        brC[i] = load_b_frag(WrT + (nc0 + i * 16) * HN, HN, lane);
      }
      for (int kt = 0; kt < HN / 32 - 1; ++kt) {
        v16bf aN = load_a_frag(&sh_a[0][0] + (kt + 1) * 32, HN, lane);
        v16bf bzN[2], brN[2];
#pragma unroll
        for (int i = 0; i < 2; ++i) {
          bzN[i] = load_b_frag(WzT + (nc0 + i * 16) * HN + (kt + 1) * 32, HN, lane);
          brN[i] = load_b_frag(WrT + (nc0 + i * 16) * HN + (kt + 1) * 32, HN, lane);
        }
#pragma unroll
        for (int i = 0; i < 2; ++i) {
          accz[i] = __builtin_amdgcn_wmma_f32_16x16x32_bf16(false, aC, false, bzC[i], (short)0, accz[i], false, false);
          accr[i] = __builtin_amdgcn_wmma_f32_16x16x32_bf16(false, aC, false, brC[i], (short)0, accr[i], false, false);
        }
        aC = aN;
#pragma unroll
        for (int i = 0; i < 2; ++i) { bzC[i] = bzN[i]; brC[i] = brN[i]; }
      }
#pragma unroll
      for (int i = 0; i < 2; ++i) {
        accz[i] = __builtin_amdgcn_wmma_f32_16x16x32_bf16(false, aC, false, bzC[i], (short)0, accz[i], false, false);
        accr[i] = __builtin_amdgcn_wmma_f32_16x16x32_bf16(false, aC, false, brC[i], (short)0, accr[i], false, false);
      }
    }
    float zf[2][8];
    __syncthreads();                                  // everyone done reading sh_a
#pragma unroll
    for (int i = 0; i < 2; ++i) {
      int ng = nc0 + i * 16 + fn;
#pragma unroll
      for (int j = 0; j < 8; ++j) {
        int m = fm + j;
        float z = 1.f / (1.f + __expf(-accz[i][j]));
        float r = 1.f / (1.f + __expf(-accr[i][j]));
        zf[i][j] = z;
        sh_a[m][ng] = f2bf(r * sh_h[m][ng]);          // stage r*h for candidate GEMM
      }
    }
    __syncthreads();

    // phase 3: candidate h_tilde GEMM (SW-pipelined) + state update
    v8f acch[2];
#pragma unroll
    for (int i = 0; i < 2; ++i) {
      int ng = nc0 + i * 16 + fn;
#pragma unroll
      for (int j = 0; j < 8; ++j) acch[i][j] = base_h[(b0 + fm + j) * HN + ng];
    }
    {
      v16bf aC = load_a_frag(&sh_a[0][0], HN, lane);
      v16bf bhC[2];
#pragma unroll
      for (int i = 0; i < 2; ++i) bhC[i] = load_b_frag(WhT + (nc0 + i * 16) * HN, HN, lane);
      for (int kt = 0; kt < HN / 32 - 1; ++kt) {
        v16bf aN = load_a_frag(&sh_a[0][0] + (kt + 1) * 32, HN, lane);
        v16bf bhN[2];
#pragma unroll
        for (int i = 0; i < 2; ++i)
          bhN[i] = load_b_frag(WhT + (nc0 + i * 16) * HN + (kt + 1) * 32, HN, lane);
#pragma unroll
        for (int i = 0; i < 2; ++i)
          acch[i] = __builtin_amdgcn_wmma_f32_16x16x32_bf16(false, aC, false, bhC[i], (short)0, acch[i], false, false);
        aC = aN;
#pragma unroll
        for (int i = 0; i < 2; ++i) bhC[i] = bhN[i];
      }
#pragma unroll
      for (int i = 0; i < 2; ++i)
        acch[i] = __builtin_amdgcn_wmma_f32_16x16x32_bf16(false, aC, false, bhC[i], (short)0, acch[i], false, false);
    }
#pragma unroll
    for (int i = 0; i < 2; ++i) {
      int ng = nc0 + i * 16 + fn;
#pragma unroll
      for (int j = 0; j < 8; ++j) {
        int m = fm + j;
        float ht = tanhf(acch[i][j]);
        float z  = zf[i][j];
        float hn = (1.f - z) * sh_h[m][ng] + z * ht;
        sh_h[m][ng] = hn;
        hs_bf[(size_t)t * BN * HN + (size_t)(b0 + m) * HN + ng] = f2bf(hn);
      }
    }
  }
}

// ---------- final projection: (T*B,H) @ (H,O) with WMMA, SW-pipelined ----------
__global__ __launch_bounds__(256) void proj_kernel(const unsigned short* __restrict__ hs_bf,
                                                   const unsigned short* __restrict__ WlT,
                                                   const float* __restrict__ blin,
                                                   float* __restrict__ out) {
  const int tid = threadIdx.x, lane = tid & 31, wave = tid >> 5;
  const int row0 = blockIdx.x * 16;                   // over T*B rows
  const int nc0  = wave * 64;
  const int fm = (lane >> 4) << 3, fn = lane & 15;
  v8f acc[4];
#pragma unroll
  for (int i = 0; i < 4; ++i) {
    float bl = blin[nc0 + i * 16 + fn];
#pragma unroll
    for (int j = 0; j < 8; ++j) acc[i][j] = bl;
  }
  const unsigned short* aBase = hs_bf + (size_t)row0 * HN;
  v16bf aC = load_a_frag(aBase, HN, lane);
  v16bf bC[4];
#pragma unroll
  for (int i = 0; i < 4; ++i) bC[i] = load_b_frag(WlT + (nc0 + i * 16) * HN, HN, lane);
  for (int kt = 0; kt < HN / 32 - 1; ++kt) {
    v16bf aN = load_a_frag(aBase + (kt + 1) * 32, HN, lane);
    v16bf bN[4];
#pragma unroll
    for (int i = 0; i < 4; ++i)
      bN[i] = load_b_frag(WlT + (nc0 + i * 16) * HN + (kt + 1) * 32, HN, lane);
#pragma unroll
    for (int i = 0; i < 4; ++i)
      acc[i] = __builtin_amdgcn_wmma_f32_16x16x32_bf16(false, aC, false, bC[i], (short)0, acc[i], false, false);
    aC = aN;
#pragma unroll
    for (int i = 0; i < 4; ++i) bC[i] = bN[i];
  }
#pragma unroll
  for (int i = 0; i < 4; ++i)
    acc[i] = __builtin_amdgcn_wmma_f32_16x16x32_bf16(false, aC, false, bC[i], (short)0, acc[i], false, false);
#pragma unroll
  for (int i = 0; i < 4; ++i) {
    int ng = nc0 + i * 16 + fn;
#pragma unroll
    for (int j = 0; j < 8; ++j)
      out[(size_t)(row0 + fm + j) * ON + ng] = acc[i][j];
  }
}

extern "C" void kernel_launch(void* const* d_in, const int* in_sizes, int n_in,
                              void* d_out, int out_size, void* d_ws, size_t ws_size,
                              hipStream_t stream) {
  const float* C    = (const float*)d_in[0];
  const float* t_in = (const float*)d_in[1];
  // d_in[2] mask: dead (forward overwrites with ones)
  const float* Wz   = (const float*)d_in[3];
  const float* bz   = (const float*)d_in[4];
  const float* Wr   = (const float*)d_in[5];
  const float* br   = (const float*)d_in[6];
  const float* Wh   = (const float*)d_in[7];
  const float* bh   = (const float*)d_in[8];
  const float* Wgh  = (const float*)d_in[9];
  const float* bgh  = (const float*)d_in[10];
  // d_in[11] wgx, d_in[12] bgx: dead (gamma_x multiplied by (1-mask)==0)
  const float* Wlin = (const float*)d_in[13];
  const float* blin = (const float*)d_in[14];
  float* out = (float*)d_out;

  char* ws = (char*)d_ws;
  unsigned short* WzT = (unsigned short*)(ws);
  unsigned short* WrT = (unsigned short*)(ws + (512ull << 10));
  unsigned short* WhT = (unsigned short*)(ws + (1024ull << 10));
  unsigned short* WlT = (unsigned short*)(ws + (1536ull << 10));
  float* fb    = (float*)(ws + (2048ull << 10));
  float* S     = fb;
  float* colz  = fb + 512;
  float* colr  = fb + 1024;
  float* colh  = fb + 1536;
  float* base_z = fb + 2048;
  float* base_r = base_z + 512 * 512;
  float* base_h = base_r + 512 * 512;
  unsigned short* hs_bf = (unsigned short*)(ws + (8192ull << 10));  // 50 MB staging

  prep_kernel<<<4104, 256, 0, stream>>>(Wz, Wr, Wh, Wgh, Wlin, bz, br, bh,
                                        WzT, WrT, WhT, WlT, S, colz, colr, colh);
  base_kernel<<<(BN * HN) / 256, 256, 0, stream>>>(C, Wz, Wr, Wh, colz, colr, colh,
                                                   base_z, base_r, base_h);
  grud_recur<<<BN / 16, 512, 0, stream>>>(t_in, bgh, S, base_z, base_r, base_h,
                                          WzT, WrT, WhT, hs_bf);
  proj_kernel<<<(T_STEPS * BN) / 16, 256, 0, stream>>>(hs_bf, WlT, blin, out);
}
